// ChannelAttentionLayer_9887014716270
// MI455X (gfx1250) — matrix operand order repaired and checked by reading, not compiled, not run on hardware
//
#include <hip/hip_runtime.h>
#include <hip/hip_bf16.h>

// ---------------------------------------------------------------------------
// Channel attention, CDNA5 (gfx1250), wave32, WMMA f16 -> f32, TDM Q-tile DMA.
//
//   B=4, C=256, H=W=64  (P=4096 query tokens)
//   K/V token grid 66x66 = 4356 tokens (1x1 conv with pad 1); BN cancels conv
//   bias, so biases are skipped entirely.
//
// Workspace budget ~88 MB (assumed available in d_ws).
// ---------------------------------------------------------------------------

typedef __attribute__((ext_vector_type(16))) _Float16 v16h;
typedef __attribute__((ext_vector_type(8)))  _Float16 v8h;
typedef __attribute__((ext_vector_type(8)))  float    v8f;
typedef __attribute__((ext_vector_type(4)))  float    v4f;
typedef __attribute__((ext_vector_type(4)))  unsigned int u32x4;
typedef __attribute__((ext_vector_type(8)))  unsigned int u32x8;

#define B_      4
#define C_      256
#define P_      4096      // 64*64 query tokens
#define KT_REAL 4356      // 66*66 key tokens
#define KT_A    4368      // 273 * 16 (key tiles for GEMM1)
#define KPAD    4384      // 137 * 32 (k-chunks for GEMM2)
#define NKT     273       // key tiles of 16
#define NKC     137       // key chunks of 32

// attention LDS layout (bytes)
#define LDS_S_BYTES   ((size_t)16 * KPAD * 4)          // 280576
#define LDS_RSUM_OFF  (LDS_S_BYTES)                    // 280576
#define LDS_STAGE_OFF (LDS_S_BYTES + 64)               // 280640 (16B aligned)
#define LDS_TOTAL     (LDS_STAGE_OFF + 16 * C_ * 2)    // 288832

__device__ __forceinline__ v8f zero8f() {
  v8f z;
#pragma unroll
  for (int i = 0; i < 8; ++i) z[i] = 0.f;
  return z;
}
__device__ __forceinline__ v16h cat8(v8h lo, v8h hi) {
  v16h r;
#pragma unroll
  for (int i = 0; i < 8; ++i) { r[i] = lo[i]; r[i + 8] = hi[i]; }
  return r;
}
__device__ __forceinline__ v8f wmma_f16(v16h a, v16h b, v8f c) {
  // D = A(16x32 f16) * B(32x16 f16) + C(16x16 f32)
  return __builtin_amdgcn_wmma_f32_16x16x32_f16(false, a, false, b,
                                                (short)0, c, false, false);
}

// ---------------------------------------------------------------------------
// Prep kernels
// ---------------------------------------------------------------------------

// Wq (OC,C,3,3) f32 -> wp[tap][oc][c] f16
__global__ __launch_bounds__(256) void pack_w3(const float* __restrict__ Wq,
                                               _Float16* __restrict__ wp) {
  int tap = blockIdx.x / C_;
  int oc  = blockIdx.x % C_;
  int c   = threadIdx.x;
  wp[((size_t)tap * C_ + oc) * C_ + c] =
      (_Float16)Wq[((size_t)oc * C_ + c) * 9 + tap];
}

// Wk/Wv (OC,C,1,1) f32 -> (oc,c) f16
__global__ __launch_bounds__(256) void pack_w1(const float* __restrict__ Wk,
                                               const float* __restrict__ Wv,
                                               _Float16* __restrict__ wk,
                                               _Float16* __restrict__ wv) {
  size_t i = (size_t)blockIdx.x * C_ + threadIdx.x;
  wk[i] = (_Float16)Wk[i];
  wv[i] = (_Float16)Wv[i];
}

// x (B,C,64,64) f32 -> xp NHWC f16 on zero-padded 66x66 grid (B,4356,C).
// Removes all boundary predication from the conv WMMA loops.
__global__ __launch_bounds__(256) void to_nhwc_pad(const float* __restrict__ x,
                                                   _Float16* __restrict__ xp) {
  int b   = blockIdx.x / KT_REAL;
  int tok = blockIdx.x % KT_REAL;
  int c   = threadIdx.x;
  int i = tok / 66, j = tok % 66;
  float v = 0.f;
  if (i >= 1 && i <= 64 && j >= 1 && j <= 64)
    v = x[((size_t)(b * C_) + c) * P_ + (i - 1) * 64 + (j - 1)];
  xp[((size_t)(b * KT_REAL) + tok) * C_ + c] = (_Float16)v;
}

// ---------------------------------------------------------------------------
// Implicit-GEMM conv3x3 via WMMA. One wave = 16 oc x 16 pixels tile.
// Block = 256 threads (8 waves, oc sub-tiles). Grid = B * 256 pixtiles * 2.
// Reads the zero-padded NHWC activation -> unconditional B loads.
// ---------------------------------------------------------------------------
__global__ __launch_bounds__(256) void conv3_wmma(const _Float16* __restrict__ xp,
                                                  const _Float16* __restrict__ wp,
                                                  float* __restrict__ qraw) {
  int wave = threadIdx.x >> 5;
  int lane = threadIdx.x & 31;
  int half = lane >> 4;
  int l16  = lane & 15;

  int blk = blockIdx.x;
  int ocg = blk & 1;  blk >>= 1;
  int pt  = blk & 255; blk >>= 8;
  int b   = blk;
  int oc0 = (ocg * 8 + wave) * 16;
  int h   = pt >> 2;
  int w0  = (pt & 3) * 16;
  int wq  = w0 + l16;  // pixel column for this lane

  v8f acc = zero8f();
  const _Float16* xbase = xp + (size_t)b * KT_REAL * C_;

#pragma unroll
  for (int tap = 0; tap < 9; ++tap) {
    int dh = tap / 3, dw = tap % 3;  // padded-grid offsets (0..2)
    const _Float16* bx = xbase + ((size_t)((h + dh) * 66 + wq + dw)) * C_ + half * 16;
    const _Float16* wa = wp + ((size_t)tap * C_ + (oc0 + l16)) * C_ + half * 16;
    __builtin_prefetch(bx + 66 * C_, 0, 1);  // next tap row
#pragma unroll
    for (int j = 0; j < 8; ++j) {
      v8h b0 = *(const v8h*)(bx + 32 * j);
      v8h b1 = *(const v8h*)(bx + 32 * j + 8);
      v8h a0 = *(const v8h*)(wa + 32 * j);
      v8h a1 = *(const v8h*)(wa + 32 * j + 8);
      acc = wmma_f16(cat8(a0, a1), cat8(b0, b1), acc);
    }
  }
  float* qp = qraw + (((size_t)b * C_ + oc0 + half * 8) * P_ + h * 64 + w0 + l16);
#pragma unroll
  for (int r = 0; r < 8; ++r) qp[(size_t)r * P_] = acc[r];
}

// ---------------------------------------------------------------------------
// Fused 1x1 conv for K and V (interior 64x64 only), WMMA, shared B loads.
// ---------------------------------------------------------------------------
__global__ __launch_bounds__(256) void conv1_wmma(const _Float16* __restrict__ xp,
                                                  const _Float16* __restrict__ wk,
                                                  const _Float16* __restrict__ wv,
                                                  float* __restrict__ kraw,
                                                  float* __restrict__ vraw) {
  int wave = threadIdx.x >> 5;
  int lane = threadIdx.x & 31;
  int half = lane >> 4;
  int l16  = lane & 15;

  int blk = blockIdx.x;
  int ocg = blk & 1;  blk >>= 1;
  int pt  = blk & 255; blk >>= 8;
  int b   = blk;
  int oc0 = (ocg * 8 + wave) * 16;
  int h   = pt >> 2;
  int w   = (pt & 3) * 16 + l16;

  v8f ak = zero8f();
  v8f av = zero8f();
  const _Float16* bx = xp + ((size_t)b * KT_REAL + (h + 1) * 66 + w + 1) * C_ + half * 16;
  const _Float16* wka = wk + (size_t)(oc0 + l16) * C_ + half * 16;
  const _Float16* wva = wv + (size_t)(oc0 + l16) * C_ + half * 16;
#pragma unroll
  for (int j = 0; j < 8; ++j) {
    v8h b0 = *(const v8h*)(bx + 32 * j);
    v8h b1 = *(const v8h*)(bx + 32 * j + 8);
    v16h bt = cat8(b0, b1);
    v8h k0 = *(const v8h*)(wka + 32 * j);
    v8h k1 = *(const v8h*)(wka + 32 * j + 8);
    v8h v0 = *(const v8h*)(wva + 32 * j);
    v8h v1 = *(const v8h*)(wva + 32 * j + 8);
    ak = wmma_f16(cat8(k0, k1), bt, ak);
    av = wmma_f16(cat8(v0, v1), bt, av);
  }
  size_t base = ((size_t)b * C_ + oc0 + half * 8) * P_ + (h * 64) + (w - l16) + l16;
#pragma unroll
  for (int r = 0; r < 8; ++r) {
    kraw[base + (size_t)r * P_] = ak[r];
    vraw[base + (size_t)r * P_] = av[r];
  }
}

// ---------------------------------------------------------------------------
// BatchNorm stats: per-channel mean/var over raw conv (stored (B,C,P_)).
// divisor may exceed stored element count (K/V zero borders). Produces
// scale = gamma*rsqrt(var+eps), shift = beta - mean*scale.
// ---------------------------------------------------------------------------
__global__ __launch_bounds__(256) void bn_stats(const float* __restrict__ raw,
                                                const float* __restrict__ gamma,
                                                const float* __restrict__ beta,
                                                float* __restrict__ scale,
                                                float* __restrict__ shift,
                                                float divisor) {
  __shared__ float s1[256], s2[256];
  int oc = blockIdx.x;
  float a = 0.f, q = 0.f;
  for (int bb = 0; bb < B_; ++bb) {
    const float* p = raw + ((size_t)bb * C_ + oc) * P_;
    for (int i = threadIdx.x; i < P_; i += 256) {
      float v = p[i];
      a += v; q += v * v;
    }
  }
  s1[threadIdx.x] = a; s2[threadIdx.x] = q;
  __syncthreads();
  for (int s = 128; s > 0; s >>= 1) {
    if (threadIdx.x < s) {
      s1[threadIdx.x] += s1[threadIdx.x + s];
      s2[threadIdx.x] += s2[threadIdx.x + s];
    }
    __syncthreads();
  }
  if (threadIdx.x == 0) {
    float mean = s1[0] / divisor;
    float var  = s2[0] / divisor - mean * mean;
    float sc   = gamma[oc] * rsqrtf(var + 1e-5f);
    scale[oc] = sc;
    shift[oc] = beta[oc] - mean * sc;
  }
}

// Q: normalize + transpose to token-major f16: qt[b][pix][c]
__global__ __launch_bounds__(256) void norm_q(const float* __restrict__ qraw,
                                              const float* __restrict__ sc,
                                              const float* __restrict__ sh,
                                              _Float16* __restrict__ qt) {
  int b   = blockIdx.x >> 12;
  int pix = blockIdx.x & 4095;
  int c   = threadIdx.x;
  float v = qraw[((size_t)b * C_ + c) * P_ + pix];
  qt[((size_t)(b * P_ + pix)) * C_ + c] = (_Float16)(v * sc[c] + sh[c]);
}

// K: token-major f16 kt[b][tok][c], tok in [0,KT_A); border=shift, pad=0
__global__ __launch_bounds__(256) void norm_k(const float* __restrict__ kraw,
                                              const float* __restrict__ sc,
                                              const float* __restrict__ sh,
                                              _Float16* __restrict__ kt) {
  int b   = blockIdx.x / KT_A;
  int tok = blockIdx.x % KT_A;
  int c   = threadIdx.x;
  float val;
  if (tok >= KT_REAL) {
    val = 0.f;
  } else {
    int i = tok / 66, j = tok % 66;
    if (i == 0 || i == 65 || j == 0 || j == 65) {
      val = sh[c];
    } else {
      val = kraw[((size_t)b * C_ + c) * P_ + (i - 1) * 64 + (j - 1)] * sc[c] + sh[c];
    }
  }
  kt[((size_t)b * KT_A + tok) * C_ + c] = (_Float16)val;
}

// V: channel-major f16 v[b][c][tok], tok in [0,KPAD); border=shift, pad=0
__global__ __launch_bounds__(256) void norm_v(const float* __restrict__ vraw,
                                              const float* __restrict__ sc,
                                              const float* __restrict__ sh,
                                              _Float16* __restrict__ vv) {
  int b = blockIdx.x >> 8;
  int c = blockIdx.x & 255;
  float scc = sc[c], shc = sh[c];
  const float* src = vraw + ((size_t)b * C_ + c) * P_;
  _Float16* dst = vv + ((size_t)b * C_ + c) * KPAD;
  for (int tok = threadIdx.x; tok < KPAD; tok += 256) {
    float val;
    if (tok >= KT_REAL) {
      val = 0.f;
    } else {
      int i = tok / 66, j = tok % 66;
      if (i == 0 || i == 65 || j == 0 || j == 65) val = shc;
      else val = src[(i - 1) * 64 + (j - 1)] * scc + shc;
    }
    dst[tok] = (_Float16)val;
  }
}

// ---------------------------------------------------------------------------
// Fused attention: one block per (batch, 16-query tile). 8 waves.
//   phase0: TDM (tensor_load_to_lds) DMAs the 16x256 f16 Q tile into LDS
//   phase1: S^T(16 q rows x KPAD keys, f32) into LDS via WMMA (K^T * Q)
//   phase2: per-row softmax; unnormalized exp re-written in place as f16
//   phase3: out = V * P via WMMA from LDS, scaled by 1/sum at store
// Dynamic LDS: 288832 bytes (CDNA5: up to 320KB/WGP).
// ---------------------------------------------------------------------------
__global__ __launch_bounds__(256) void attn_wmma(const _Float16* __restrict__ qt,
                                                 const _Float16* __restrict__ kt,
                                                 const _Float16* __restrict__ vv,
                                                 float* __restrict__ out) {
  extern __shared__ char smem[];
  float* S    = (float*)smem;                      // [16][KPAD] f32 rows
  float* rsum = (float*)(smem + LDS_RSUM_OFF);     // [16]

  int wave = threadIdx.x >> 5;
  int lane = threadIdx.x & 31;
  int half = lane >> 4;
  int l16  = lane & 15;

  int b  = blockIdx.x >> 8;
  int q0 = (blockIdx.x & 255) * 16;

  // --- phase 0: TDM DMA of Q tile (16 rows x 256 c, f16) into LDS stage ---
  if (wave == 0) {
    unsigned long long ga =
        (unsigned long long)(const void*)(qt + ((size_t)(b * P_ + q0)) * C_);
    u32x4 g0;
    g0[0] = 1u;                                        // count=1, user mode
    g0[1] = (unsigned)LDS_STAGE_OFF;                   // lds_addr (bytes)
    g0[2] = (unsigned)(ga & 0xffffffffu);              // global_addr[31:0]
    g0[3] = (unsigned)((ga >> 32) & 0x1ffffffu)        // global_addr[56:32]
            | (2u << 30);                              // type = 2 (image)
    u32x8 g1;
    g1[0] = (1u << 16);                                // data_size = 2 bytes
    g1[1] = ((unsigned)C_ & 0xffffu) << 16;            // tensor_dim0 lo16 = 256
    g1[2] = (((unsigned)C_ >> 16) & 0xffffu)           // tensor_dim0 hi16
            | ((16u & 0xffffu) << 16);                 // tensor_dim1 lo16 = 16
    g1[3] = ((unsigned)C_ << 16);                      // tensor_dim1 hi | tile_dim0=256
    g1[4] = 16u;                                       // tile_dim1=16, tile_dim2=0
    g1[5] = (unsigned)C_;                              // tensor_dim0_stride lo32
    g1[6] = 0u;                                        // stride0 hi | stride1 lo
    g1[7] = 0u;                                        // stride1 hi
    asm volatile("tensor_load_to_lds %0, %1" : : "s"(g0), "s"(g1) : "memory");
    __builtin_amdgcn_s_wait_tensorcnt(0);
  }
  __syncthreads();

  // --- load Q tile as B-operand registers from LDS stage ---
  v16h Bq[8];
  {
    const _Float16* qls =
        (const _Float16*)(smem + LDS_STAGE_OFF) + (l16 * C_ + half * 16);
#pragma unroll
    for (int j = 0; j < 8; ++j) {
      v8h lo = *(const v8h*)(qls + 32 * j);
      v8h hi = *(const v8h*)(qls + 32 * j + 8);
      Bq[j] = cat8(lo, hi);
    }
  }

  // --- phase 1: scores ---
  for (int t = wave; t < NKT; t += 8) {
    v8f acc = zero8f();
    const _Float16* ka = kt + ((size_t)b * KT_A + t * 16 + l16) * C_ + half * 16;
    __builtin_prefetch(ka + 8 * 16 * C_, 0, 1);  // next key tile for this wave
#pragma unroll
    for (int j = 0; j < 8; ++j) {
      v8h a0 = *(const v8h*)(ka + 32 * j);
      v8h a1 = *(const v8h*)(ka + 32 * j + 8);
      acc = wmma_f16(cat8(a0, a1), Bq[j], acc);
    }
    // S^T: row q = l16; keys t*16 + half*8 + r (8 contiguous f32)
    float* sp = S + (size_t)l16 * KPAD + t * 16 + half * 8;
    *(v4f*)sp       = __builtin_shufflevector(acc, acc, 0, 1, 2, 3);
    *(v4f*)(sp + 4) = __builtin_shufflevector(acc, acc, 4, 5, 6, 7);
  }
  __syncthreads();

  // mask padded keys to -inf
  for (int i = threadIdx.x; i < 16 * (KPAD - KT_REAL); i += 256) {
    int q = i / (KPAD - KT_REAL);
    int k = KT_REAL + i % (KPAD - KT_REAL);
    S[(size_t)q * KPAD + k] = -3.0e38f;
  }
  __syncthreads();

  // --- phase 2: softmax per q row (wave handles rows wave, wave+8) ---
#pragma unroll
  for (int rr = 0; rr < 2; ++rr) {
    int q = wave + rr * 8;
    float* row = S + (size_t)q * KPAD;
    float m = -3.0e38f;
    for (int k = lane; k < KPAD; k += 32) m = fmaxf(m, row[k]);
#pragma unroll
    for (int off = 16; off > 0; off >>= 1) m = fmaxf(m, __shfl_xor(m, off, 32));
    float s = 0.f;
    _Float16* prow = (_Float16*)row;  // overlay f16 exp values over the row
    for (int k = lane; k < KPAD; k += 32) {
      float e = __expf(row[k] - m);   // load (all lanes) precedes store
      s += e;
      prow[k] = (_Float16)e;          // writes only touch lower f32 slots
    }
#pragma unroll
    for (int off = 16; off > 0; off >>= 1) s += __shfl_xor(s, off, 32);
    if (lane == 0) rsum[q] = 1.0f / s;
  }
  __syncthreads();

  // --- phase 3: out = V * P (wave handles c-tiles wave, wave+8) ---
  float rs = rsum[l16];
  for (int ct = wave; ct < 16; ct += 8) {
    int c0 = ct * 16;
    v8f acc = zero8f();
    const _Float16* va = vv + ((size_t)b * C_ + c0 + l16) * KPAD + half * 16;
    const _Float16* pb = (const _Float16*)(S + (size_t)l16 * KPAD) + half * 16;
#pragma unroll 4
    for (int ch = 0; ch < NKC; ++ch) {
      v8h a0 = *(const v8h*)(va + 32 * ch);
      v8h a1 = *(const v8h*)(va + 32 * ch + 8);
      v8h b0 = *(const v8h*)(pb + 32 * ch);
      v8h b1 = *(const v8h*)(pb + 32 * ch + 8);
      acc = wmma_f16(cat8(a0, a1), cat8(b0, b1), acc);
    }
    float* op = out + (((size_t)b * C_ + c0 + half * 8) * P_ + q0 + l16);
#pragma unroll
    for (int r = 0; r < 8; ++r) op[(size_t)r * P_] = acc[r] * rs;
  }
}

// ---------------------------------------------------------------------------
extern "C" void kernel_launch(void* const* d_in, const int* in_sizes, int n_in,
                              void* d_out, int out_size, void* d_ws, size_t ws_size,
                              hipStream_t stream) {
  (void)in_sizes; (void)n_in; (void)out_size; (void)ws_size;

  const float* x     = (const float*)d_in[0];
  const float* Wq    = (const float*)d_in[1];
  const float* gq    = (const float*)d_in[3];
  const float* betaq = (const float*)d_in[4];
  const float* Wk    = (const float*)d_in[5];
  const float* gk    = (const float*)d_in[7];
  const float* betak = (const float*)d_in[8];
  const float* Wv    = (const float*)d_in[9];
  const float* gv    = (const float*)d_in[11];
  const float* betav = (const float*)d_in[12];
  float* out = (float*)d_out;

  // workspace carve-up (256B aligned)
  char* base = (char*)d_ws;
  size_t o = 0;
  auto carve = [&](size_t bytes) -> char* {
    char* p = base + o;
    o += (bytes + 255) & ~(size_t)255;
    return p;
  };
  _Float16* xp    = (_Float16*)carve((size_t)B_ * KT_REAL * C_ * 2);  // padded NHWC x
  _Float16* wqp   = (_Float16*)carve((size_t)9 * C_ * C_ * 2);        // packed Wq
  _Float16* wkp   = (_Float16*)carve((size_t)C_ * C_ * 2);
  _Float16* wvp   = (_Float16*)carve((size_t)C_ * C_ * 2);
  float*    qraw  = (float*)carve((size_t)B_ * C_ * P_ * 4);
  float*    kraw  = (float*)carve((size_t)B_ * C_ * P_ * 4);
  float*    vraw  = (float*)carve((size_t)B_ * C_ * P_ * 4);
  float*    scQ   = (float*)carve(C_ * 4);
  float*    shQ   = (float*)carve(C_ * 4);
  float*    scK   = (float*)carve(C_ * 4);
  float*    shK   = (float*)carve(C_ * 4);
  float*    scV   = (float*)carve(C_ * 4);
  float*    shV   = (float*)carve(C_ * 4);
  _Float16* qtb   = (_Float16*)carve((size_t)B_ * P_ * C_ * 2);       // (b,pix,c)
  _Float16* ktb   = (_Float16*)carve((size_t)B_ * KT_A * C_ * 2);     // (b,tok,c)
  _Float16* vb    = (_Float16*)carve((size_t)B_ * C_ * KPAD * 2);     // (b,c,tok)

  // 1) pack / transpose (zero-padded 66x66 NHWC removes conv border masking)
  pack_w3<<<9 * C_, 256, 0, stream>>>(Wq, wqp);
  pack_w1<<<C_, 256, 0, stream>>>(Wk, Wv, wkp, wvp);
  to_nhwc_pad<<<B_ * KT_REAL, 256, 0, stream>>>(x, xp);

  // 2) convs (implicit GEMM, WMMA)
  conv3_wmma<<<B_ * 256 * 2, 256, 0, stream>>>(xp, wqp, qraw);
  conv1_wmma<<<B_ * 256 * 2, 256, 0, stream>>>(xp, wkp, wvp, kraw, vraw);

  // 3) BN statistics (bias cancels in BN -> skipped)
  bn_stats<<<C_, 256, 0, stream>>>(qraw, gq, betaq, scQ, shQ, (float)(B_ * P_));
  bn_stats<<<C_, 256, 0, stream>>>(kraw, gk, betak, scK, shK, (float)(B_ * KT_REAL));
  bn_stats<<<C_, 256, 0, stream>>>(vraw, gv, betav, scV, shV, (float)(B_ * KT_REAL));

  // 4) normalize + convert to f16 in GEMM-friendly layouts
  norm_q<<<B_ * P_, 256, 0, stream>>>(qraw, scQ, shQ, qtb);
  norm_k<<<B_ * KT_A, 256, 0, stream>>>(kraw, scK, shK, ktb);
  norm_v<<<B_ * C_, 256, 0, stream>>>(vraw, scV, shV, vb);

  // 5) fused attention (TDM Q-tile DMA; scores + softmax in LDS; WMMA GEMMs)
  attn_wmma<<<B_ * 256, 256, LDS_TOTAL, stream>>>(qtb, ktb, vb, out);
}
